// QEffDeepseekV3Attention_83700322664922
// MI455X (gfx1250) — matrix operand order repaired and checked by reading, not compiled, not run on hardware
//
#include <hip/hip_runtime.h>
#include <stddef.h>

// ---------------------------------------------------------------------------
// CDNA5 (gfx1250) wave32 WMMA attention pipeline, bf16 operands / f32 accum.
// ---------------------------------------------------------------------------
typedef __bf16 bf16;
typedef __attribute__((ext_vector_type(16))) __bf16 v16bf;
typedef __attribute__((ext_vector_type(8)))  __bf16 v8bf;
typedef __attribute__((ext_vector_type(8)))  float  v8f;
typedef int v4i __attribute__((vector_size(4 * sizeof(int))));

// ---- async global->LDS copy (ASYNCcnt path), guarded so compile can't fail -
#if defined(__AMDGCN__) && __has_builtin(__builtin_amdgcn_global_load_async_to_lds_b128)
#define ASYNC_LDS 1
#else
#define ASYNC_LDS 0
#endif

#define S_    2048
#define NH_   16
#define QL_   1536
#define KVL_  512
#define ROPE_ 64
#define VD_   128
#define CAT_  576          // KVL + ROPE
#define SCALE_ 0.07216878364870322f   // (128+64)^-0.5
#define LN1E4_OVER_32 0.28782313662425575f

static __device__ __forceinline__ void async_cp16(void* lds_dst, const void* gsrc) {
#if ASYNC_LDS
    __builtin_amdgcn_global_load_async_to_lds_b128(
        (__attribute__((address_space(1))) v4i*)(gsrc),
        (__attribute__((address_space(3))) v4i*)(lds_dst), 0, 0);
#else
    *(v8bf*)lds_dst = *(const v8bf*)gsrc;
#endif
}
static __device__ __forceinline__ void async_wait() {
#if ASYNC_LDS
#if __has_builtin(__builtin_amdgcn_s_wait_asynccnt)
    __builtin_amdgcn_s_wait_asynccnt(0);
#else
    asm volatile("s_wait_asynccnt 0" ::: "memory");
#endif
#endif
}

static __device__ __forceinline__ v8f wmma_bf16(v16bf a, v16bf b, v8f c) {
    return __builtin_amdgcn_wmma_f32_16x16x32_bf16(false, a, false, b, (short)0, c, false, false);
}

// A-fragment (16x32 bf16, MxK): lane<16 holds K 0..7 & 16..23; lane>=16 holds K 8..15 & 24..31
static __device__ __forceinline__ v16bf frag_a(const bf16* base, int lane) {
    const int koff = (lane < 16) ? 0 : 8;
    v8bf lo = *(const v8bf*)(base + koff);
    v8bf hi = *(const v8bf*)(base + koff + 16);
    v16bf r;
#pragma unroll
    for (int e = 0; e < 8; ++e) { r[e] = lo[e]; r[e + 8] = hi[e]; }
    return r;
}

// B-fragment (32x16 bf16, KxN): lane<16 holds K 0..7 & 16..23 of col N=lane; lane>=16 holds K 8..15 & 24..31
static __device__ __forceinline__ v16bf frag_b(const bf16* base, int lane) {
    const int koff = (lane < 16) ? 0 : 8;
    v8bf lo = *(const v8bf*)(base + koff);
    v8bf hi = *(const v8bf*)(base + koff + 16);
    v16bf r;
#pragma unroll
    for (int e = 0; e < 8; ++e) { r[e] = lo[e]; r[e + 8] = hi[e]; }
    return r;
}

static __device__ __forceinline__ void store_c(float* C, size_t i, float v) { C[i] = v; }
static __device__ __forceinline__ void store_c(bf16*  C, size_t i, float v) { C[i] = (bf16)v; }

// A-tile loaders: 128 rows x 64 cols into LDS (row stride 72 bf16).
static __device__ __forceinline__ void ld_a_tile(const float* A, bf16* As, int tid, int m0, int k0, int lda) {
#pragma unroll
    for (int i = 0; i < 8; ++i) {
        const int idx = tid + i * 256;
        const int row = idx >> 4, c4 = idx & 15;
        const float4 vv = *(const float4*)(A + (size_t)(m0 + row) * lda + k0 + c4 * 4);
        bf16* p = &As[row * 72 + c4 * 4];
        p[0] = (bf16)vv.x; p[1] = (bf16)vv.y; p[2] = (bf16)vv.z; p[3] = (bf16)vv.w;
    }
}
static __device__ __forceinline__ void ld_a_tile(const bf16* A, bf16* As, int tid, int m0, int k0, int lda) {
#pragma unroll
    for (int i = 0; i < 4; ++i) {
        const int idx = tid + i * 256;
        const int row = idx >> 3, c8 = idx & 7;
        async_cp16(&As[row * 72 + c8 * 8], A + (size_t)(m0 + row) * lda + k0 + c8 * 8);
    }
}

// ---------------------------------------------------------------------------
// Generic bf16-WMMA GEMM: C = A[M,K] * B (+scale), f32 accumulate.
//   B_NK = true : B stored (N,K) row-major ("A @ B^T"); false: B stored (K,N)
//   CT   = true : store transposed, C[col*ldc + row]
// Block tile 128x64, K-step 64; 8 waves 4x2, wave = 2x2 WMMA tiles, 8 WMMA/K-step.
// ---------------------------------------------------------------------------
template <typename AT, bool B_NK, bool CT, typename OT>
__global__ __launch_bounds__(256) void gemm_wmma(
    const AT* __restrict__ A, const float* __restrict__ B, OT* __restrict__ C,
    int M, int N, int K, int lda, int ldb, int ldc,
    long long bStrideZ, long long cStrideZ, float scale)
{
    (void)M; (void)N;
    __shared__ bf16 As[128 * 72];
    __shared__ bf16 Bs[64 * 72];

    const int tid  = threadIdx.x;
    const int lane = tid & 31;
    const int wid  = tid >> 5;
    const int wm   = wid >> 1, wn = wid & 1;
    const int m0   = blockIdx.y * 128;
    const int n0   = blockIdx.x * 64;

    B += (size_t)blockIdx.z * bStrideZ;
    C += (size_t)blockIdx.z * cStrideZ;

    v8f acc[2][2] = {};

    for (int k0 = 0; k0 < K; k0 += 64) {
        ld_a_tile(A, As, tid, m0, k0, lda);
        if (B_NK) {
#pragma unroll
            for (int i = 0; i < 4; ++i) {
                const int idx = tid + i * 256;
                const int row = idx >> 4, c4 = idx & 15;
                const float4 vv = *(const float4*)(B + (size_t)(n0 + row) * ldb + k0 + c4 * 4);
                bf16* p = &Bs[row * 72 + c4 * 4];
                p[0] = (bf16)vv.x; p[1] = (bf16)vv.y; p[2] = (bf16)vv.z; p[3] = (bf16)vv.w;
            }
        } else {
#pragma unroll
            for (int i = 0; i < 16; ++i) {
                const int idx = tid + i * 256;
                const int kk = idx >> 6, nn = idx & 63;
                Bs[nn * 72 + kk] = (bf16)B[(size_t)(k0 + kk) * ldb + n0 + nn];
            }
        }
        if (k0 + 64 < K)
            __builtin_prefetch(A + (size_t)(m0 + (tid >> 2)) * lda + k0 + 64, 0, 0);
        async_wait();
        __syncthreads();

#pragma unroll
        for (int kk = 0; kk < 64; kk += 32) {
            v16bf af[2], bfr[2];
#pragma unroll
            for (int mi = 0; mi < 2; ++mi)
                af[mi] = frag_a(&As[(wm * 32 + mi * 16 + (lane & 15)) * 72 + kk], lane);
#pragma unroll
            for (int ni = 0; ni < 2; ++ni)
                bfr[ni] = frag_b(&Bs[(wn * 32 + ni * 16 + (lane & 15)) * 72 + kk], lane);
#pragma unroll
            for (int mi = 0; mi < 2; ++mi)
#pragma unroll
                for (int ni = 0; ni < 2; ++ni)
                    acc[mi][ni] = wmma_bf16(af[mi], bfr[ni], acc[mi][ni]);
        }
        __syncthreads();
    }

#pragma unroll
    for (int mi = 0; mi < 2; ++mi)
#pragma unroll
        for (int ni = 0; ni < 2; ++ni) {
            const int col = n0 + wn * 32 + ni * 16 + (lane & 15);
#pragma unroll
            for (int j = 0; j < 8; ++j) {
                const int row = m0 + wm * 32 + mi * 16 + j + ((lane >> 4) << 3);
                const size_t idx = CT ? ((size_t)col * ldc + row) : ((size_t)row * ldc + col);
                store_c(C, idx, acc[mi][ni][j] * scale);
            }
        }
}

// ---------------------------------------------------------------------------
// RMSNorm(ckv) + RoPE(k_pe) -> kv_cat bf16 (S, 576)
// ---------------------------------------------------------------------------
__global__ __launch_bounds__(256) void kv_norm_rope(
    const float* __restrict__ ckv_full, const float* __restrict__ kvw,
    const int* __restrict__ pos, bf16* __restrict__ kvcat)
{
    __shared__ float red[256];
    __shared__ float srms;
    const int s = blockIdx.x, tid = threadIdx.x;
    const float* row = ckv_full + (size_t)s * CAT_;
    float acc = 0.f;
    for (int c = tid; c < KVL_; c += 256) { const float x = row[c]; acc += x * x; }
    red[tid] = acc; __syncthreads();
    for (int off = 128; off > 0; off >>= 1) { if (tid < off) red[tid] += red[tid + off]; __syncthreads(); }
    if (tid == 0) srms = rsqrtf(red[0] * (1.0f / (float)KVL_) + 1e-6f);
    __syncthreads();
    const float rms = srms;
    bf16* out = kvcat + (size_t)s * CAT_;
    for (int c = tid; c < KVL_; c += 256) out[c] = (bf16)(kvw[c] * row[c] * rms);
    if (tid < 32) {
        const int j = tid;
        const float t = (float)pos[s];
        const float inv = __expf(-(float)j * LN1E4_OVER_32);
        float cs, sn; __sincosf(t * inv, &cs, &sn);
        const float x0 = row[KVL_ + 2 * j], x1 = row[KVL_ + 2 * j + 1];
        out[KVL_ + j]      = (bf16)(x0 * cs - x1 * sn);
        out[KVL_ + 32 + j] = (bf16)(x1 * cs + x0 * sn);
    }
}

// RMSNorm(q_a) f32 -> bf16 (S, 1536)
__global__ __launch_bounds__(256) void q_norm(
    const float* __restrict__ qa_f, const float* __restrict__ qw, bf16* __restrict__ qa_b)
{
    __shared__ float red[256];
    __shared__ float srms;
    const int s = blockIdx.x, tid = threadIdx.x;
    const float* row = qa_f + (size_t)s * QL_;
    float acc = 0.f;
    for (int c = tid; c < QL_; c += 256) { const float x = row[c]; acc += x * x; }
    red[tid] = acc; __syncthreads();
    for (int off = 128; off > 0; off >>= 1) { if (tid < off) red[tid] += red[tid + off]; __syncthreads(); }
    if (tid == 0) srms = rsqrtf(red[0] * (1.0f / (float)QL_) + 1e-6f);
    __syncthreads();
    const float rms = srms;
    bf16* o = qa_b + (size_t)s * QL_;
    for (int c = tid; c < QL_; c += 256) o[c] = (bf16)(qw[c] * row[c] * rms);
}

// RoPE(q_pe) f32 (S, NH*64) -> q_cat[h][s][512..575] bf16, SCALE folded in.
__global__ __launch_bounds__(32) void qpe_rope(
    const float* __restrict__ qpe, const int* __restrict__ pos, bf16* __restrict__ qcat, int S)
{
    const int s = blockIdx.x, h = blockIdx.y, j = threadIdx.x;
    const float t = (float)pos[s];
    const float inv = __expf(-(float)j * LN1E4_OVER_32);
    float cs, sn; __sincosf(t * inv, &cs, &sn);
    const float* src = qpe + (size_t)s * (NH_ * ROPE_) + h * ROPE_;
    bf16* dst = qcat + ((size_t)h * S + s) * CAT_ + KVL_;
    const float x0 = src[2 * j], x1 = src[2 * j + 1];
    dst[j]      = (bf16)(SCALE_ * (x0 * cs - x1 * sn));
    dst[32 + j] = (bf16)(SCALE_ * (x1 * cs + x0 * sn));
}

// ---------------------------------------------------------------------------
// Flash attention per (64-row tile, head). Q resident in LDS (async-loaded);
// K streamed 64x64 double-buffered via async-to-LDS; V^T tiles async-loaded;
// online softmax (4 threads/row) with reference-compatible -10000 mask.
// ---------------------------------------------------------------------------
__global__ __launch_bounds__(256) void attn_flash(
    const bf16* __restrict__ qcat, const bf16* __restrict__ kvcat,
    const bf16* __restrict__ vT, bf16* __restrict__ obuf, int S)
{
    extern __shared__ char smem[];
    bf16*  Qs   = (bf16*)smem;               // 64 x 584
    bf16*  Ks   = Qs + 64 * 584;             // 2 x (64 x 72) double buffer
    bf16*  Vs   = Ks + 2 * 64 * 72;          // 128 x 72 ([v][t])
    bf16*  Ps   = Vs + 128 * 72;             // 64 x 72
    float* sc   = (float*)(Ps + 64 * 72);    // 64 x 66
    float* corr = sc + 64 * 66;              // 64
    float* lsum = corr + 64;                 // 64
    float* rowm = lsum + 64;                 // 64
    float* pm   = rowm + 64;                 // 256 (partials)

    const int tid  = threadIdx.x;
    const int lane = tid & 31;
    const int wid  = tid >> 5;
    const int wm   = wid >> 1;               // row group (4)
    const int wn   = wid & 1;                // key-col half for scores
    const int wv   = wid & 1;                // VD half for P@V
    const int h    = blockIdx.y;
    const int rb   = blockIdx.x * 64;
    const bf16* qh = qcat + (size_t)h * S * CAT_;

    auto load_ks = [&](bf16* dst, int kb, int kc) {
#pragma unroll
        for (int i = 0; i < 2; ++i) {
            const int idx = tid + i * 256;
            const int row = idx >> 3, c8 = idx & 7;
            async_cp16(&dst[row * 72 + c8 * 8],
                       kvcat + (size_t)(kb * 64 + row) * CAT_ + kc * 64 + c8 * 8);
        }
    };

    // Q tile: 64 x 576 async into LDS
#pragma unroll
    for (int i = 0; i < 18; ++i) {
        const int idx = tid + i * 256;
        const int row = idx / 72, c8 = idx % 72;
        async_cp16(&Qs[row * 584 + c8 * 8], qh + (size_t)(rb + row) * CAT_ + c8 * 8);
    }
    v8f o_acc[4] = {};
    float m_i = -1e30f, l_i = 0.0f;
    async_wait();
    __syncthreads();

    for (int kb = 0; kb < S / 64; ++kb) {
        v8f s_acc[2] = {};
        // ---- scores: stream K in 64x64 chunks, double-buffered async
        load_ks(Ks, kb, 0);
        async_wait();
        __syncthreads();
        for (int kc = 0; kc < 9; ++kc) {
            bf16* cur = Ks + (kc & 1) * (64 * 72);
            if (kc + 1 < 9) load_ks(Ks + ((kc + 1) & 1) * (64 * 72), kb, kc + 1);
#pragma unroll
            for (int kk = 0; kk < 2; ++kk) {
                const v16bf a = frag_a(&Qs[(wm * 16 + (lane & 15)) * 584 + kc * 64 + kk * 32], lane);
#pragma unroll
                for (int ni = 0; ni < 2; ++ni) {
                    const v16bf b = frag_b(&cur[(wn * 32 + ni * 16 + (lane & 15)) * 72 + kk * 32], lane);
                    s_acc[ni] = wmma_bf16(a, b, s_acc[ni]);
                }
            }
            async_wait();
            __syncthreads();
        }
        // ---- masked scores -> LDS (SCALE already folded into q_cat)
#pragma unroll
        for (int ni = 0; ni < 2; ++ni) {
            const int col = wn * 32 + ni * 16 + (lane & 15);
#pragma unroll
            for (int j = 0; j < 8; ++j) {
                const int row = wm * 16 + j + ((lane >> 4) << 3);
                float vv = s_acc[ni][j];
                if (kb * 64 + col > rb + row) vv = -10000.0f;
                sc[row * 66 + col] = vv;
            }
        }
        __syncthreads();
        // ---- issue V^T tile async copies early; they hide under the softmax
#pragma unroll
        for (int i = 0; i < 4; ++i) {
            const int idx = tid + i * 256;
            const int v = idx >> 3, t8 = idx & 7;
            async_cp16(&Vs[v * 72 + t8 * 8],
                       vT + (size_t)(h * VD_ + v) * S + kb * 64 + t8 * 8);
        }
        // ---- online softmax: 4 threads per row
        const int r = tid >> 2, q4 = tid & 3;
        {
            float mx = -1e30f;
#pragma unroll
            for (int j = q4 * 16; j < q4 * 16 + 16; ++j) mx = fmaxf(mx, sc[r * 66 + j]);
            pm[tid] = mx;
        }
        __syncthreads();
        if (tid < 64) {
            const float bm = fmaxf(fmaxf(pm[tid * 4], pm[tid * 4 + 1]),
                                   fmaxf(pm[tid * 4 + 2], pm[tid * 4 + 3]));
            const float nm = fmaxf(m_i, bm);
            corr[tid] = __expf(m_i - nm);
            rowm[tid] = nm;
            m_i = nm;
        }
        __syncthreads();
        {
            const float nm = rowm[r];
            float ssum = 0.f;
#pragma unroll
            for (int j = q4 * 16; j < q4 * 16 + 16; ++j) {
                const float p = __expf(sc[r * 66 + j] - nm);
                Ps[r * 72 + j] = (bf16)p;
                ssum += p;
            }
            pm[tid] = ssum;
        }
        __syncthreads();
        if (tid < 64)
            l_i = l_i * corr[tid] + (pm[tid * 4] + pm[tid * 4 + 1] + pm[tid * 4 + 2] + pm[tid * 4 + 3]);
        async_wait();          // V^T tile complete
        __syncthreads();
        // ---- rescale O, then accumulate P@V
#pragma unroll
        for (int nt = 0; nt < 4; ++nt)
#pragma unroll
            for (int j = 0; j < 8; ++j) {
                const int row = wm * 16 + j + ((lane >> 4) << 3);
                o_acc[nt][j] *= corr[row];
            }
#pragma unroll
        for (int kt = 0; kt < 2; ++kt) {
            const v16bf a = frag_a(&Ps[(wm * 16 + (lane & 15)) * 72 + kt * 32], lane);
#pragma unroll
            for (int nt = 0; nt < 4; ++nt) {
                const v16bf b = frag_b(&Vs[(wv * 64 + nt * 16 + (lane & 15)) * 72 + kt * 32], lane);
                o_acc[nt] = wmma_bf16(a, b, o_acc[nt]);
            }
        }
        __syncthreads();
    }

    if (tid < 64) lsum[tid] = fmaxf(l_i, 1e-20f);
    __syncthreads();
#pragma unroll
    for (int nt = 0; nt < 4; ++nt) {
        const int col = h * VD_ + wv * 64 + nt * 16 + (lane & 15);
#pragma unroll
        for (int j = 0; j < 8; ++j) {
            const int row = wm * 16 + j + ((lane >> 4) << 3);
            obuf[(size_t)(rb + row) * (NH_ * VD_) + col] = (bf16)(o_acc[nt][j] / lsum[row]);
        }
    }
}

// ---------------------------------------------------------------------------
extern "C" void kernel_launch(void* const* d_in, const int* in_sizes, int n_in,
                              void* d_out, int out_size, void* d_ws, size_t ws_size,
                              hipStream_t stream)
{
    (void)in_sizes; (void)n_in; (void)out_size; (void)ws_size;
    const float* hidden   = (const float*)d_in[0];
    const float* W_q_a    = (const float*)d_in[1];
    const float* W_kv_a   = (const float*)d_in[2];
    const float* q_a_nw   = (const float*)d_in[3];
    const float* kv_a_nw  = (const float*)d_in[4];
    const float* q_rope_w = (const float*)d_in[5];
    const float* fusedqk  = (const float*)d_in[6];
    const float* v_up     = (const float*)d_in[7];
    const float* W_o      = (const float*)d_in[8];
    // d_in[9]: attention_mask (exactly causal -> applied analytically)
    const int*   pos      = (const int*)d_in[10];
    float*       out      = (float*)d_out;

    char* ws = (char*)d_ws;
    size_t off = 0;
    auto carve = [&](size_t bytes) -> void* {
        void* p = ws + off;
        off = (off + bytes + 255) & ~(size_t)255;
        return p;
    };
    float* ckv_full = (float*)carve((size_t)S_ * CAT_ * 4);
    float* qa_f     = (float*)carve((size_t)S_ * QL_  * 4);
    float* qpe_f    = (float*)carve((size_t)S_ * NH_ * ROPE_ * 4);
    bf16*  qa_b     = (bf16*) carve((size_t)S_ * QL_  * 2);
    bf16*  kvcat    = (bf16*) carve((size_t)S_ * CAT_ * 2);
    bf16*  qcat     = (bf16*) carve((size_t)NH_ * S_ * CAT_ * 2);
    bf16*  vT       = (bf16*) carve((size_t)NH_ * VD_ * S_ * 2);
    bf16*  abuf     = (bf16*) carve((size_t)S_ * NH_ * VD_ * 2);

    // 1) ckv_full = hidden @ W_kv_a^T   (f32 out, pre-norm)
    gemm_wmma<float, true, false, float><<<dim3(CAT_ / 64, S_ / 128, 1), 256, 0, stream>>>(
        hidden, W_kv_a, ckv_full, S_, CAT_, 2048, 2048, 2048, CAT_, 0, 0, 1.0f);
    // 2) qa_f = hidden @ W_q_a^T
    gemm_wmma<float, true, false, float><<<dim3(QL_ / 64, S_ / 128, 1), 256, 0, stream>>>(
        hidden, W_q_a, qa_f, S_, QL_, 2048, 2048, 2048, QL_, 0, 0, 1.0f);
    // 3) kv_cat = [rmsnorm(ckv), rope(k_pe)]  (bf16)
    kv_norm_rope<<<S_, 256, 0, stream>>>(ckv_full, kv_a_nw, pos, kvcat);
    // 4) qa_b = rmsnorm(qa_f) (bf16)
    q_norm<<<S_, 256, 0, stream>>>(qa_f, q_a_nw, qa_b);
    // 5) qpe_f = qa_b @ q_rope_w
    gemm_wmma<bf16, false, false, float><<<dim3((NH_ * ROPE_) / 64, S_ / 128, 1), 256, 0, stream>>>(
        qa_b, q_rope_w, qpe_f, S_, NH_ * ROPE_, QL_, QL_, NH_ * ROPE_, NH_ * ROPE_, 0, 0, 1.0f);
    // 6) rope(q_pe) -> q_cat[...,512:576], SCALE folded
    qpe_rope<<<dim3(S_, NH_), 32, 0, stream>>>(qpe_f, pos, qcat, S_);
    // 7) q_abs[h] = qa_b @ fusedqk[h] -> q_cat[h][...,0:512], SCALE folded (z = head)
    gemm_wmma<bf16, false, false, bf16><<<dim3(KVL_ / 64, S_ / 128, NH_), 256, 0, stream>>>(
        qa_b, fusedqk, qcat, S_, KVL_, QL_, QL_, KVL_, CAT_,
        (long long)QL_ * KVL_, (long long)S_ * CAT_, SCALE_);
    // 8) vT = (ckv(bf16, lda=576) @ v_up) stored transposed: vT[n][s]
    gemm_wmma<bf16, false, true, bf16><<<dim3((NH_ * VD_) / 64, S_ / 128, 1), 256, 0, stream>>>(
        kvcat, v_up, vT, S_, NH_ * VD_, KVL_, CAT_, NH_ * VD_, /*ldc=*/S_, 0, 0, 1.0f);
    // 9) flash attention per (64-row tile, head)
    const size_t smem_bytes = (size_t)(64 * 584 + 2 * 64 * 72 + 128 * 72 + 64 * 72) * 2
                            + (size_t)(64 * 66 + 64 + 64 + 64 + 256) * 4;   // ~140 KB
    attn_flash<<<dim3(S_ / 64, NH_), 256, smem_bytes, stream>>>(qcat, kvcat, vT, abuf, S_);
    // 10) out = abuf @ W_o^T  (f32 out)
    gemm_wmma<bf16, true, false, float><<<dim3(2048 / 64, S_ / 128, 1), 256, 0, stream>>>(
        abuf, W_o, out, S_, 2048, NH_ * VD_, NH_ * VD_, NH_ * VD_, 2048, 0, 0, 1.0f);
}
